// MultiheadAttention_68642167324915
// MI455X (gfx1250) — matrix operand order repaired and checked by reading, not compiled, not run on hardware
//
#include <hip/hip_runtime.h>

typedef __attribute__((ext_vector_type(16))) _Float16 v16h;
typedef __attribute__((ext_vector_type(8)))  float    v8f;
typedef __attribute__((ext_vector_type(4)))  float    f32x4;
typedef __attribute__((ext_vector_type(4)))  unsigned int u32x4;
typedef __attribute__((ext_vector_type(8)))  int      i32x8;
typedef __attribute__((ext_vector_type(4)))  int      i32x4;

#define DEV static __device__ __forceinline__

constexpr int BATCH = 2;
constexpr int SEQ   = 2048;
constexpr int NH    = 12;
constexpr int DKH   = 64;            // head dim
constexpr int DM    = 768;           // model dim
constexpr int MTOT  = BATCH * SEQ;   // 4096 flattened rows
constexpr int KBLK  = 32;            // key block per attention step

// ---------------------------------------------------------------------------
// WMMA wrapper: D = A(16x32 f16) * B(32x16 f16) + C(16x16 f32)
// ---------------------------------------------------------------------------
DEV v8f wmma16(v16h a, v16h b, v8f c) {
  return __builtin_amdgcn_wmma_f32_16x16x32_f16(
      /*neg_a=*/false, a, /*neg_b=*/false, b,
      /*c_mod=*/(short)0, c, /*reuse_a=*/false, /*reuse_b=*/false);
}

// ---------------------------------------------------------------------------
// Fragment loaders (this code's contiguous k order for A/B, wave32)
// A 16x32 f16: lane holds row M=lane%16; halves 0..15 -> K=k0+(lane/16)*16+i
//              (contiguous, matching B)
// B 32x16 f16: lane holds col N=lane%16; lanes<16: K=k0..k0+15 contiguous,
//              lanes>=16: K=k0+16..k0+31
// C/D 16x16 f32: elem e -> row M=e+8*(lane/16), col N=lane%16
// ---------------------------------------------------------------------------
DEV v16h loadA_f16(const _Float16* __restrict__ src, int ld, int row0, int k0, int lane) {
  const int r = lane & 15, hh = lane >> 4;
  const _Float16* p = src + (size_t)(row0 + r) * ld + (k0 + hh * 16);
  v16h a;
#pragma unroll
  for (int i = 0; i < 8; ++i) a[i] = p[i];
#pragma unroll
  for (int i = 0; i < 8; ++i) a[8 + i] = p[8 + i];
  return a;
}

// B = (row-major src)^T tile: B[k][n] = S[n0+n][k0+k]; contiguous along k.
DEV v16h loadBT_f16(const _Float16* __restrict__ S, int ld, int n0, int k0, int lane) {
  const int n = lane & 15, hh = lane >> 4;
  const _Float16* p = S + (size_t)(n0 + n) * ld + (k0 + hh * 16);
  v16h b;
#pragma unroll
  for (int i = 0; i < 16; ++i) b[i] = p[i];
  return b;
}

// Same pattern but reading an LDS tile (rowstride in elements).
DEV v16h ldsBT(const _Float16* tile, int rowstride, int n0, int k0, int lane) {
  const int n = lane & 15, hh = lane >> 4;
  const _Float16* p = tile + (n0 + n) * rowstride + (k0 + hh * 16);
  v16h b;
#pragma unroll
  for (int i = 0; i < 16; ++i) b[i] = p[i];
  return b;
}

// half-wave (16-lane) reductions: one C-tile row lives across a 16-lane group
DEV float hmax16(float v) {
#pragma unroll
  for (int off = 1; off < 16; off <<= 1) v = fmaxf(v, __shfl_xor(v, off, 32));
  return v;
}
DEV float hsum16(float v) {
#pragma unroll
  for (int off = 1; off < 16; off <<= 1) v += __shfl_xor(v, off, 32);
  return v;
}

// ---------------------------------------------------------------------------
// TDM: 2-D tile (data_size = 2 bytes) global -> LDS.  D# packing per
// CDNA5 ISA (group0: count/lds/global/type,
// group1: data_size, tensor dims, tile dims, stride; groups 2/3 zero = 2-D).
// 6-arg builtin form (clang-23 / therock headers).
// ---------------------------------------------------------------------------
DEV void tdm_load_2d(unsigned lds_byte_off, const void* gptr,
                     unsigned tile_d0, unsigned tile_d1,
                     unsigned tensor_d0, unsigned tensor_d1,
                     unsigned stride0_elems) {
  const unsigned long long ga = (unsigned long long)gptr;
  u32x4 g0;
  g0[0] = 1u;                                    // count=1 (valid), is_restore=0
  g0[1] = lds_byte_off;                          // lds_addr   [63:32]
  g0[2] = (unsigned)ga;                          // global_addr[95:64]
  g0[3] = (unsigned)((ga >> 32) & 0x01ffffffu)   // global_addr[120:96]
          | (2u << 30);                          // type=2 ("image")
  i32x8 g1;
  g1[0] = (int)(1u << 16);                       // data_size=1 -> 2-byte elems
  g1[1] = (int)((tensor_d0 & 0xffffu) << 16);    // tensor_dim0[15:0] @ bits 63:48
  g1[2] = (int)(((tensor_d0 >> 16) & 0xffffu) |  // tensor_dim0[31:16]
                ((tensor_d1 & 0xffffu) << 16));  // tensor_dim1[15:0]
  g1[3] = (int)(((tensor_d1 >> 16) & 0xffffu) |  // tensor_dim1[31:16]
                ((tile_d0 & 0xffffu) << 16));    // tile_dim0
  g1[4] = (int)(tile_d1 & 0xffffu);              // tile_dim1 (tile_dim2=0)
  g1[5] = (int)stride0_elems;                    // tensor_dim0_stride[31:0]
  g1[6] = 0;                                     // stride0[47:32], stride1[15:0]
  g1[7] = 0;
  const i32x4 z4 = {0, 0, 0, 0};
  const i32x8 z8 = {0, 0, 0, 0, 0, 0, 0, 0};
  __builtin_amdgcn_tensor_load_to_lds(g0, g1, z4, z4, z8, 0);
}

DEV unsigned lds_off(const void* p) {            // flat LDS addr[31:0] == LDS offset
  return (unsigned)(size_t)p;
}

// ---------------------------------------------------------------------------
// Kernel 0: f32 -> f16 convert (4 elems / thread)
// ---------------------------------------------------------------------------
__global__ __launch_bounds__(256)
void cvt_kernel(const float* __restrict__ src, _Float16* __restrict__ dst) {
  const size_t i = ((size_t)blockIdx.x * 256 + threadIdx.x) * 4;
  const f32x4 v = *(const f32x4*)(src + i);
  _Float16 h0 = (_Float16)v.x, h1 = (_Float16)v.y;
  _Float16 h2 = (_Float16)v.z, h3 = (_Float16)v.w;
  dst[i + 0] = h0; dst[i + 1] = h1; dst[i + 2] = h2; dst[i + 3] = h3;
}

// ---------------------------------------------------------------------------
// Kernel 1: projection  Yh = X @ W^T + b -> head-major f16 [B,H,S,DK]
// (used for Q and K). 8 waves/block; wave w computes a 32x64 output tile
// (two A row-tiles share the four B fragments -> 8 WMMAs / k-step).
// ---------------------------------------------------------------------------
__global__ __launch_bounds__(256)
void proj_kernel(const _Float16* __restrict__ X, const _Float16* __restrict__ W,
                 const float* __restrict__ bias, _Float16* __restrict__ Yh) {
  const int lane = threadIdx.x & 31;
  const int wave = threadIdx.x >> 5;
  const int m0 = blockIdx.y * 256 + wave * 32;
  const int n0 = blockIdx.x * 64;

  v8f acc[2][4] = {};
  for (int k0 = 0; k0 < DM; k0 += 32) {
    __builtin_prefetch(X + (size_t)(m0 + (lane & 15)) * DM + k0 + 64, 0, 3);
    v16h a0 = loadA_f16(X, DM, m0, k0, lane);
    v16h a1 = loadA_f16(X, DM, m0 + 16, k0, lane);
#pragma unroll
    for (int t = 0; t < 4; ++t) {
      v16h b = loadBT_f16(W, DM, n0 + t * 16, k0, lane);
      acc[0][t] = wmma16(a0, b, acc[0][t]);
      acc[1][t] = wmma16(a1, b, acc[1][t]);
    }
  }

  const int hh = lane >> 4, cn = lane & 15;
#pragma unroll
  for (int rt = 0; rt < 2; ++rt) {
#pragma unroll
    for (int t = 0; t < 4; ++t) {
      const int n  = n0 + t * 16 + cn;
      const int hd = n / DKH, dk = n % DKH;
      const float bb = bias[n];
#pragma unroll
      for (int e = 0; e < 8; ++e) {
        const int m  = m0 + rt * 16 + e + 8 * hh;
        const int bi = m >> 11;           // / SEQ
        const int s  = m & (SEQ - 1);     // % SEQ
        Yh[(((size_t)bi * NH + hd) * SEQ + s) * DKH + dk] = (_Float16)(acc[rt][t][e] + bb);
      }
    }
  }
}

// ---------------------------------------------------------------------------
// Kernel 2: transposed projection  Vt = (X @ Wv^T)^T = Wv @ X^T,
// stored [B,H,DK,S] f16 so attention PV B-fragments are contiguous.
// Wave computes 32 feature-rows x 64 data-rows; stores coalesce along S.
// ---------------------------------------------------------------------------
__global__ __launch_bounds__(256)
void projT_kernel(const _Float16* __restrict__ X, const _Float16* __restrict__ W,
                  const float* __restrict__ bias, _Float16* __restrict__ Vt) {
  const int lane = threadIdx.x & 31;
  const int wave = threadIdx.x >> 5;
  const int f0 = blockIdx.y * 256 + wave * 32;   // feature rows (0..767)
  const int n0 = blockIdx.x * 64;                // data rows   (0..4095)

  v8f acc[2][4] = {};
  for (int k0 = 0; k0 < DM; k0 += 32) {
    __builtin_prefetch(X + (size_t)(n0 + (lane & 15)) * DM + k0 + 64, 0, 3);
    v16h a0 = loadA_f16(W, DM, f0, k0, lane);
    v16h a1 = loadA_f16(W, DM, f0 + 16, k0, lane);
#pragma unroll
    for (int t = 0; t < 4; ++t) {
      v16h b = loadBT_f16(X, DM, n0 + t * 16, k0, lane);
      acc[0][t] = wmma16(a0, b, acc[0][t]);
      acc[1][t] = wmma16(a1, b, acc[1][t]);
    }
  }

  const int hh = lane >> 4, cn = lane & 15;
#pragma unroll
  for (int rt = 0; rt < 2; ++rt) {
#pragma unroll
    for (int e = 0; e < 8; ++e) {
      const int f  = f0 + rt * 16 + e + 8 * hh;
      const int hd = f / DKH, dk = f % DKH;
      const float bb = bias[f];
#pragma unroll
      for (int t = 0; t < 4; ++t) {
        const int m  = n0 + t * 16 + cn;
        const int bi = m >> 11;
        const int s  = m & (SEQ - 1);
        Vt[(((size_t)bi * NH + hd) * DKH + dk) * SEQ + s] = (_Float16)(acc[rt][t][e] + bb);
      }
    }
  }
}

// ---------------------------------------------------------------------------
// Kernel 3: flash attention per (b,h). 8 waves/block, wave owns 16 q-rows.
// K tile (32x64) and V^T tile (64x32) staged into LDS by the Tensor Data
// Mover, double-buffered; online softmax; output merged heads f16 [B,S,D].
// ---------------------------------------------------------------------------
__global__ __launch_bounds__(256)
void attn_kernel(const _Float16* __restrict__ Qh, const _Float16* __restrict__ Kh,
                 const _Float16* __restrict__ Vt, _Float16* __restrict__ Oh) {
  __shared__ _Float16 ldsK[2][KBLK * DKH];   // [32 keys][64 dk]   4KB each
  __shared__ _Float16 ldsV[2][DKH * KBLK];   // [64 dk][32 keys]   4KB each
  __shared__ _Float16 ldsP[8][16 * KBLK];    // per-wave P tiles   8KB

  const int lane = threadIdx.x & 31;
  const int wave = threadIdx.x >> 5;
  const int b = blockIdx.z;
  const int h = blockIdx.y;
  const int q0 = blockIdx.x * 128 + wave * 16;

  const size_t head = ((size_t)b * NH + h) * SEQ * DKH;
  const _Float16* Q  = Qh + head;
  const _Float16* K  = Kh + head;
  const _Float16* VT = Vt + head;    // [DK][S] row-major per head

  // Q fragments for this wave's 16 rows (dk 0..31, 32..63), kept in regs
  const v16h qa0 = loadA_f16(Q, DKH, q0, 0, lane);
  const v16h qa1 = loadA_f16(Q, DKH, q0, 32, lane);

  v8f o0 = {}, o1 = {}, o2 = {}, o3 = {};
  float m_[8], l_[8];
#pragma unroll
  for (int e = 0; e < 8; ++e) { m_[e] = -1.0e30f; l_[e] = 0.0f; }

  const float scale = 0.125f;        // 1/sqrt(DK)
  const int hh = lane >> 4, cn = lane & 15;
  _Float16* P = ldsP[wave];

  // prologue: TDM-stage tile 0 into buffer 0 (one wave issues; TENSORcnt is
  // per-wave, so the issuing wave waits, then a workgroup barrier publishes)
  if (wave == 0) {
    tdm_load_2d(lds_off(&ldsK[0][0]), K,  DKH, KBLK, DKH, SEQ, DKH);
    tdm_load_2d(lds_off(&ldsV[0][0]), VT, KBLK, DKH, SEQ, DKH, SEQ);
  }

  for (int kb = 0; kb < SEQ; kb += KBLK) {
    const int cur = (kb >> 5) & 1;
    if (wave == 0) __builtin_amdgcn_s_wait_tensorcnt((short)0);
    __syncthreads();                          // tile[cur] visible to all waves

    if (wave == 0 && (kb + KBLK) < SEQ) {     // kick off next tile into buf[1-cur]
      tdm_load_2d(lds_off(&ldsK[1 - cur][0]), K + (size_t)(kb + KBLK) * DKH,
                  DKH, KBLK, DKH, SEQ, DKH);
      tdm_load_2d(lds_off(&ldsV[1 - cur][0]), VT + (kb + KBLK),
                  KBLK, DKH, SEQ, DKH, SEQ);
    }

    // ---- scores: S = Q (16x64) @ K^T (64x32) from LDS K tile
    v8f s0 = {}, s1 = {};
    s0 = wmma16(qa0, ldsBT(ldsK[cur], DKH,  0,  0, lane), s0);
    s0 = wmma16(qa1, ldsBT(ldsK[cur], DKH,  0, 32, lane), s0);
    s1 = wmma16(qa0, ldsBT(ldsK[cur], DKH, 16,  0, lane), s1);
    s1 = wmma16(qa1, ldsBT(ldsK[cur], DKH, 16, 32, lane), s1);

    // ---- online softmax (row stats replicated across each 16-lane group)
    float alpha[8];
#pragma unroll
    for (int e = 0; e < 8; ++e) {
      const float a0 = s0[e] * scale;
      const float a1 = s1[e] * scale;
      const float tmax = hmax16(fmaxf(a0, a1));
      const float nm = fmaxf(m_[e], tmax);
      alpha[e] = __expf(m_[e] - nm);
      m_[e] = nm;
      const float p0 = __expf(a0 - nm);
      const float p1 = __expf(a1 - nm);
      s0[e] = p0;
      s1[e] = p1;
      l_[e] = l_[e] * alpha[e] + hsum16(p0 + p1);
    }
#pragma unroll
    for (int e = 0; e < 8; ++e) {
      o0[e] *= alpha[e]; o1[e] *= alpha[e];
      o2[e] *= alpha[e]; o3[e] *= alpha[e];
    }

    // ---- C-layout -> A-layout for P via wave-private LDS tile
#pragma unroll
    for (int e = 0; e < 8; ++e) {
      P[(e + 8 * hh) * KBLK + cn]      = (_Float16)s0[e];
      P[(e + 8 * hh) * KBLK + 16 + cn] = (_Float16)s1[e];
    }
    v16h pa;
    {
      const _Float16* pr = P + (size_t)cn * KBLK + hh * 16;
#pragma unroll
      for (int i = 0; i < 8; ++i) pa[i] = pr[i];
#pragma unroll
      for (int i = 0; i < 8; ++i) pa[8 + i] = pr[8 + i];
    }

    // ---- O += P (16x32) @ V (32x64) from LDS V^T tile (contiguous rows)
    o0 = wmma16(pa, ldsBT(ldsV[cur], KBLK,  0, 0, lane), o0);
    o1 = wmma16(pa, ldsBT(ldsV[cur], KBLK, 16, 0, lane), o1);
    o2 = wmma16(pa, ldsBT(ldsV[cur], KBLK, 32, 0, lane), o2);
    o3 = wmma16(pa, ldsBT(ldsV[cur], KBLK, 48, 0, lane), o3);

    __syncthreads();                          // done reading buf[cur]
  }

  // ---- epilogue: normalize and store merged heads [B,S,D] f16
  v8f oo[4] = {o0, o1, o2, o3};
#pragma unroll
  for (int t = 0; t < 4; ++t) {
    const int dk = t * 16 + cn;
#pragma unroll
    for (int e = 0; e < 8; ++e) {
      const int qr = q0 + e + 8 * hh;
      Oh[((size_t)b * SEQ + qr) * DM + h * DKH + dk] = (_Float16)(oo[t][e] / l_[e]);
    }
  }
}

// ---------------------------------------------------------------------------
// Kernel 4: output projection  out = O @ Wo^T + bo  (f16 in, f32 out),
// 32x64 output tile per wave like proj_kernel.
// ---------------------------------------------------------------------------
__global__ __launch_bounds__(256)
void outproj_kernel(const _Float16* __restrict__ Oh, const _Float16* __restrict__ Wo,
                    const float* __restrict__ bo, float* __restrict__ Y) {
  const int lane = threadIdx.x & 31;
  const int wave = threadIdx.x >> 5;
  const int m0 = blockIdx.y * 256 + wave * 32;
  const int n0 = blockIdx.x * 64;

  v8f acc[2][4] = {};
  for (int k0 = 0; k0 < DM; k0 += 32) {
    __builtin_prefetch(Oh + (size_t)(m0 + (lane & 15)) * DM + k0 + 64, 0, 3);
    v16h a0 = loadA_f16(Oh, DM, m0, k0, lane);
    v16h a1 = loadA_f16(Oh, DM, m0 + 16, k0, lane);
#pragma unroll
    for (int t = 0; t < 4; ++t) {
      v16h b = loadBT_f16(Wo, DM, n0 + t * 16, k0, lane);
      acc[0][t] = wmma16(a0, b, acc[0][t]);
      acc[1][t] = wmma16(a1, b, acc[1][t]);
    }
  }

  const int hh = lane >> 4, cn = lane & 15;
#pragma unroll
  for (int rt = 0; rt < 2; ++rt) {
#pragma unroll
    for (int t = 0; t < 4; ++t) {
      const int n = n0 + t * 16 + cn;
      const float bb = bo[n];
#pragma unroll
      for (int e = 0; e < 8; ++e) {
        const int m = m0 + rt * 16 + e + 8 * hh;
        Y[(size_t)m * DM + n] = acc[rt][t][e] + bb;
      }
    }
  }
}

// ---------------------------------------------------------------------------
extern "C" void kernel_launch(void* const* d_in, const int* in_sizes, int n_in,
                              void* d_out, int out_size, void* d_ws, size_t ws_size,
                              hipStream_t stream) {
  (void)in_sizes; (void)n_in; (void)out_size; (void)ws_size;

  const float* q  = (const float*)d_in[0];
  const float* k  = (const float*)d_in[1];
  const float* v  = (const float*)d_in[2];
  // d_in[3] = mask: all-ones, a no-op in the reference
  const float* Wq = (const float*)d_in[4];
  const float* bq = (const float*)d_in[5];
  const float* Wk = (const float*)d_in[6];
  const float* bk = (const float*)d_in[7];
  const float* Wv = (const float*)d_in[8];
  const float* bv = (const float*)d_in[9];
  const float* Wo = (const float*)d_in[10];
  const float* bo = (const float*)d_in[11];
  float* out = (float*)d_out;

  constexpr size_t NX = (size_t)MTOT * DM;   // 3,145,728
  constexpr size_t NW = (size_t)DM * DM;     //   589,824

  _Float16* ws = (_Float16*)d_ws;
  _Float16* Xq  = ws;                 // f16 copies of activations
  _Float16* Xk  = Xq  + NX;
  _Float16* Xv  = Xk  + NX;
  _Float16* Wqh = Xv  + NX;           // f16 copies of weights
  _Float16* Wkh = Wqh + NW;
  _Float16* Wvh = Wkh + NW;
  _Float16* Woh = Wvh + NW;
  _Float16* Qh  = Woh + NW;           // head-major [B,H,S,DK]
  _Float16* Kh  = Qh  + NX;
  _Float16* Vt  = Kh  + NX;           // transposed [B,H,DK,S]
  _Float16* Oh  = Vt  + NX;           // merged heads [B,S,D]

  const dim3 blk(256);

  // f32 -> f16 converts (1024 elems / block)
  cvt_kernel<<<dim3(NX / 1024), blk, 0, stream>>>(q,  Xq);
  cvt_kernel<<<dim3(NX / 1024), blk, 0, stream>>>(k,  Xk);
  cvt_kernel<<<dim3(NX / 1024), blk, 0, stream>>>(v,  Xv);
  cvt_kernel<<<dim3(NW / 1024), blk, 0, stream>>>(Wq, Wqh);
  cvt_kernel<<<dim3(NW / 1024), blk, 0, stream>>>(Wk, Wkh);
  cvt_kernel<<<dim3(NW / 1024), blk, 0, stream>>>(Wv, Wvh);
  cvt_kernel<<<dim3(NW / 1024), blk, 0, stream>>>(Wo, Woh);

  // projections
  const dim3 gproj(DM / 64, MTOT / 256);          // (12, 16)
  proj_kernel<<<gproj, blk, 0, stream>>>(Xq, Wqh, bq, Qh);
  proj_kernel<<<gproj, blk, 0, stream>>>(Xk, Wkh, bk, Kh);
  const dim3 gprojT(MTOT / 64, DM / 256);         // (64, 3)
  projT_kernel<<<gprojT, blk, 0, stream>>>(Xv, Wvh, bv, Vt);

  // attention
  const dim3 gattn(SEQ / 128, NH, BATCH);         // (16, 12, 2)
  attn_kernel<<<gattn, blk, 0, stream>>>(Qh, Kh, Vt, Oh);

  // output projection
  const dim3 gout(DM / 64, MTOT / 256);           // (12, 16)
  outproj_kernel<<<gout, blk, 0, stream>>>(Oh, Woh, bo, out);
}
